// SwinTransformerBlock_18262200943508
// MI455X (gfx1250) — matrix operand order repaired and checked by reading, not compiled, not run on hardware
//
#include <hip/hip_runtime.h>
#include <math.h>
#include <stdint.h>

// ---------------------------------------------------------------------------
// Swin block for MI455X (gfx1250): bf16 WMMA (v_wmma_f32_16x16x32_bf16) with
// fp32 accumulation; GEMMs use double-buffered GLOBAL_LOAD_ASYNC_TO_LDS_B128
// staging (ASYNCcnt-tracked, VGPR-free) per CDNA5 ISA.
// ---------------------------------------------------------------------------

typedef unsigned short u16;
typedef unsigned long long u64;
typedef __attribute__((ext_vector_type(16))) __bf16 v16bf;
typedef __attribute__((ext_vector_type(8)))  float  v8f;

#define WMMA_BF16(a, b, c) \
  __builtin_amdgcn_wmma_f32_16x16x32_bf16(false, (a), false, (b), (short)0, (c), false, false)

// Problem constants
#define DIMC   384
#define HEADS  12
#define HD     32
#define NTOK   64          // tokens per window
#define BB     32
#define HH     64
#define WW     64
#define NWIN   64          // windows per image (8x8)
#define TOKS   (BB*HH*WW)  // 131072
#define MLPH   (4*DIMC)    // 1536
#define QKVN   (3*DIMC)    // 1152

#if __has_builtin(__builtin_amdgcn_s_wait_asynccnt)
#define WAIT_ASYNC(n) __builtin_amdgcn_s_wait_asynccnt(n)
#else
__device__ __forceinline__ void __wait_async_impl(int n) {
  if (n == 0) asm volatile("s_wait_asynccnt 0x0" ::: "memory");
  else        asm volatile("s_wait_asynccnt 0x4" ::: "memory");
}
#define WAIT_ASYNC(n) __wait_async_impl(n)
#endif

// Two async b128 loads: 32 contiguous bytes global -> 32 contiguous bytes LDS.
// INST_OFFSET applies to both LDS and global address, so offset:16 covers the
// second half with the same base registers.
__device__ __forceinline__ void async_ld32B(unsigned lds_off, u64 gaddr) {
  asm volatile("global_load_async_to_lds_b128 %0, %1, off\n\t"
               "global_load_async_to_lds_b128 %0, %1, off offset:16"
               :: "v"(lds_off), "v"(gaddr) : "memory");
}

__device__ __forceinline__ u16 f2bf(float f) {
  union { float f; unsigned u; } v; v.f = f;
  unsigned u = v.u;
  u += 0x7fffu + ((u >> 16) & 1u);   // round-to-nearest-even
  return (u16)(u >> 16);
}

__device__ __forceinline__ v16bf ldfrag2(const u16* p0, const u16* p1) {
  v16bf r;
  ((uint4*)&r)[0] = *(const uint4*)p0;
  ((uint4*)&r)[1] = *(const uint4*)p1;
  return r;
}
__device__ __forceinline__ v16bf ldfrag1(const u16* p) {
  v16bf r;
  ((uint4*)&r)[0] = ((const uint4*)p)[0];
  ((uint4*)&r)[1] = ((const uint4*)p)[1];
  return r;
}

// ---------------------------------------------------------------------------
// Weight convert + transpose: src (K x N) fp32 row-major -> dst (N x K) bf16
// ---------------------------------------------------------------------------
__global__ __launch_bounds__(256)
void wcvt_t(const float* __restrict__ src, u16* __restrict__ dst, int K, int N) {
  int idx = blockIdx.x * 256 + threadIdx.x;
  if (idx < K * N) {
    int k = idx / N, n = idx % N;
    dst[(size_t)n * K + k] = f2bf(src[idx]);
  }
}

// ---------------------------------------------------------------------------
// LN1 + cyclic shift (-4,-4) + window partition -> bf16 tokens (win-ordered)
// ---------------------------------------------------------------------------
__global__ __launch_bounds__(256)
void ln_shift_window(const float* __restrict__ x, const float* __restrict__ g,
                     const float* __restrict__ b, u16* __restrict__ out) {
  int wv = threadIdx.x >> 5, lane = threadIdx.x & 31;
  int tok = blockIdx.x * 8 + wv;            // output token, window-ordered
  int win = tok >> 6, n = tok & 63;
  int bimg = win >> 6, wimg = win & 63;
  int y  = (((wimg >> 3) << 3) + (n >> 3) + 4) & 63;   // roll(-SHIFT)
  int xx = (((wimg & 7) << 3) + (n & 7) + 4) & 63;
  const float* src = x + (((size_t)bimg * HH + y) * WW + xx) * DIMC;

  float v[12]; float s = 0.f, s2 = 0.f;
#pragma unroll
  for (int i = 0; i < 12; ++i) { v[i] = src[lane + i * 32]; s += v[i]; s2 += v[i] * v[i]; }
#pragma unroll
  for (int m = 1; m < 32; m <<= 1) { s += __shfl_xor(s, m, 32); s2 += __shfl_xor(s2, m, 32); }
  float mean = s * (1.0f / DIMC);
  float var  = s2 * (1.0f / DIMC) - mean * mean;
  float inv  = rsqrtf(var + 1e-5f);

  u16* dst = out + (size_t)tok * DIMC;
#pragma unroll
  for (int i = 0; i < 12; ++i) {
    int c = lane + i * 32;
    dst[c] = f2bf((v[i] - mean) * inv * g[c] + b[c]);
  }
}

// ---------------------------------------------------------------------------
// Window-reverse + un-shift + residual + LN2 (fused)
// ---------------------------------------------------------------------------
__global__ __launch_bounds__(256)
void residual_ln(const float* __restrict__ x, const float* __restrict__ ptmp,
                 const float* __restrict__ g, const float* __restrict__ b,
                 float* __restrict__ x2, u16* __restrict__ t2) {
  int wv = threadIdx.x >> 5, lane = threadIdx.x & 31;
  int tok = blockIdx.x * 8 + wv;            // flat (b, y, x) token
  int bimg = tok >> 12;
  int y  = (tok >> 6) & 63;
  int xx = tok & 63;
  int sy = (y + 60) & 63, sx = (xx + 60) & 63;       // roll(+SHIFT) inverse
  int win = bimg * NWIN + ((sy >> 3) << 3) + (sx >> 3);
  int n   = ((sy & 7) << 3) + (sx & 7);
  const float* pr = ptmp + ((size_t)win * NTOK + n) * DIMC;
  const float* px = x + (size_t)tok * DIMC;
  float* pox = x2 + (size_t)tok * DIMC;

  float v[12]; float s = 0.f, s2 = 0.f;
#pragma unroll
  for (int i = 0; i < 12; ++i) {
    int c = lane + i * 32;
    v[i] = px[c] + pr[c];
    pox[c] = v[i];
    s += v[i]; s2 += v[i] * v[i];
  }
#pragma unroll
  for (int m = 1; m < 32; m <<= 1) { s += __shfl_xor(s, m, 32); s2 += __shfl_xor(s2, m, 32); }
  float mean = s * (1.0f / DIMC);
  float var  = s2 * (1.0f / DIMC) - mean * mean;
  float inv  = rsqrtf(var + 1e-5f);
  u16* dst = t2 + (size_t)tok * DIMC;
#pragma unroll
  for (int i = 0; i < 12; ++i) {
    int c = lane + i * 32;
    dst[c] = f2bf((v[i] - mean) * inv * g[c] + b[c]);
  }
}

// ---------------------------------------------------------------------------
// bf16 WMMA GEMM:  A (M x K) bf16  x  Bt (N x K) bf16  + bias
// Block tile 64x64, 4 waves, wave tile 32x32 (2x2 x v_wmma_f32_16x16x32_bf16).
// Double-buffered LDS staged with GLOBAL_LOAD_ASYNC_TO_LDS_B128 (ASYNCcnt).
// EPI: 0 = bf16 out, 1 = fp32 out, 2 = GELU->bf16, 3 = +residual -> fp32
// ---------------------------------------------------------------------------
template <int EPI>
__global__ __launch_bounds__(128)
void gemm_bf16(const u16* __restrict__ A, const u16* __restrict__ Bt,
               const float* __restrict__ bias, const float* __restrict__ res,
               u16* __restrict__ outH, float* __restrict__ outF,
               int M, int N, int K) {
  __shared__ u16 sA[2][64 * 32];
  __shared__ u16 sB[2][64 * 32];
  int tid = threadIdx.x;
  int wv = tid >> 5, lane = tid & 31;
  int l15 = lane & 15, lh = lane >> 4;
  int nB = blockIdx.x * 64, mB = blockIdx.y * 64;
  int wr = (wv >> 1) * 32, wc = (wv & 1) * 32;

  v8f acc[2][2] = {};

  int srow = tid >> 1;                 // 0..63
  int shalf = (tid & 1) * 16;          // 0 or 16
  const u16* gA = A + (size_t)(mB + srow) * K + shalf;
  const u16* gB = Bt + (size_t)(nB + srow) * K + shalf;
  unsigned ldsA[2] = { (unsigned)(uintptr_t)&sA[0][srow * 32 + shalf],
                       (unsigned)(uintptr_t)&sA[1][srow * 32 + shalf] };
  unsigned ldsB[2] = { (unsigned)(uintptr_t)&sB[0][srow * 32 + shalf],
                       (unsigned)(uintptr_t)&sB[1][srow * 32 + shalf] };

  // prologue: stage k-step 0 into buffer 0 (4 async b128 per thread)
  async_ld32B(ldsA[0], (u64)(uintptr_t)gA);
  async_ld32B(ldsB[0], (u64)(uintptr_t)gB);

  int nk = K >> 5;
  int kb = lh * 8;
  for (int i = 0; i < nk; ++i) {
    int cur = i & 1;
    if (i + 1 < nk) {
      // prefetch next k-step into the other buffer while this one computes
      async_ld32B(ldsA[cur ^ 1], (u64)(uintptr_t)(gA + ((i + 1) << 5)));
      async_ld32B(ldsB[cur ^ 1], (u64)(uintptr_t)(gB + ((i + 1) << 5)));
      WAIT_ASYNC(4);   // in-order: first 4 outstanding (= stage i) retired
    } else {
      WAIT_ASYNC(0);
    }
    __syncthreads();

    v16bf a0 = ldfrag1(&sA[cur][(wr + l15) * 32 + lh * 16]);
    v16bf a1 = ldfrag1(&sA[cur][(wr + 16 + l15) * 32 + lh * 16]);
    v16bf b0 = ldfrag1(&sB[cur][(wc + l15) * 32 + lh * 16]);
    v16bf b1 = ldfrag1(&sB[cur][(wc + 16 + l15) * 32 + lh * 16]);

    acc[0][0] = WMMA_BF16(a0, b0, acc[0][0]);
    acc[0][1] = WMMA_BF16(a0, b1, acc[0][1]);
    acc[1][0] = WMMA_BF16(a1, b0, acc[1][0]);
    acc[1][1] = WMMA_BF16(a1, b1, acc[1][1]);

    if (i + 1 < nk) __syncthreads();   // readers done before next overwrite
  }

#pragma unroll
  for (int ti = 0; ti < 2; ++ti)
#pragma unroll
    for (int tj = 0; tj < 2; ++tj) {
      const float* av = (const float*)&acc[ti][tj];
#pragma unroll
      for (int r = 0; r < 8; ++r) {
        int row = mB + wr + ti * 16 + r + (lh << 3);
        int col = nB + wc + tj * 16 + l15;
        float v = av[r] + bias[col];
        size_t o = (size_t)row * N + col;
        if (EPI == 0) {
          outH[o] = f2bf(v);
        } else if (EPI == 1) {
          outF[o] = v;
        } else if (EPI == 2) {
          float gl = 0.5f * v * (1.0f + erff(v * 0.70710678118654752f));
          outH[o] = f2bf(gl);
        } else {
          outF[o] = v + res[o];
        }
      }
    }
}

// ---------------------------------------------------------------------------
// Windowed attention, one block per (window, head), 4 waves.
// scores = (q @ k^T) * scale + rel_bias + shift_mask ; softmax ; @ v
// ---------------------------------------------------------------------------
__global__ __launch_bounds__(128)
void attn_kernel(const u16* __restrict__ qkv, const float* __restrict__ btab,
                 u16* __restrict__ outH) {
  __shared__ u16 vT[HD * NTOK];        // v transposed: [c][t]
  __shared__ u16 sAttn[4 * 16 * NTOK]; // softmaxed scores bf16, per wave

  int blk = blockIdx.x;
  int win = blk / HEADS, h = blk % HEADS;
  int wimg = win & (NWIN - 1);
  int wy = (wimg >> 3) << 3, wx = (wimg & 7) << 3;

  int tid = threadIdx.x;
  int wv = tid >> 5, lane = tid & 31;
  int l15 = lane & 15, lh = lane >> 4;
  const float scale = 0.17677669529663687f;  // 1/sqrt(32)

  const u16* base = qkv + (size_t)win * NTOK * QKVN;

  // ---- stage v transposed into LDS ----
  {
    int t = tid >> 1;                 // token 0..63
    int c0 = (tid & 1) * 16;          // col half
    const u16* pv = base + t * QKVN + 2 * DIMC + h * HD + c0;
    u16 tmp[16];
    *(uint4*)&tmp[0] = *(const uint4*)pv;
    *(uint4*)&tmp[8] = *(const uint4*)(pv + 8);
#pragma unroll
    for (int i = 0; i < 16; ++i) vT[(c0 + i) * NTOK + t] = tmp[i];
  }
  __syncthreads();

  // ---- scores: q(16x32) @ k^T tiles ----
  int kb = lh * 8;
  int qrow = wv * 16 + l15;
  const u16* qp = base + qrow * QKVN + h * HD;
  v16bf aq = ldfrag1(qp + lh * 16);

  v8f sc[4];
  v8f zc = {};
#pragma unroll
  for (int tj = 0; tj < 4; ++tj) {
    const u16* kp = base + (tj * 16 + l15) * QKVN + DIMC + h * HD + lh * 16;
    v16bf bk = ldfrag1(kp);
    sc[tj] = WMMA_BF16(aq, bk, zc);
  }

  // ---- scale + rel-pos bias + shift mask ----
  float sv[4][8];
#pragma unroll
  for (int tj = 0; tj < 4; ++tj) {
    const float* f = (const float*)&sc[tj];
#pragma unroll
    for (int r = 0; r < 8; ++r) {
      int rm = wv * 16 + r + (lh << 3);
      int cn = tj * 16 + l15;
      int ry = rm >> 3, rx = rm & 7, cy = cn >> 3, cx = cn & 7;
      int rel = (ry - cy + 7) * 15 + (rx - cx + 7);
      float bias = btab[rel * HEADS + h];
      int ty = wy + ry, tx = wx + rx, uy = wy + cy, ux = wx + cx;
      int idr = (ty < 56 ? 0 : (ty < 60 ? 1 : 2)) * 3 + (tx < 56 ? 0 : (tx < 60 ? 1 : 2));
      int idc = (uy < 56 ? 0 : (uy < 60 ? 1 : 2)) * 3 + (ux < 56 ? 0 : (ux < 60 ? 1 : 2));
      float mk = (idr != idc) ? -100.0f : 0.0f;
      sv[tj][r] = f[r] * scale + bias + mk;
    }
  }

  // ---- softmax over 64 cols per row ----
#pragma unroll
  for (int r = 0; r < 8; ++r) {
    float mx = fmaxf(fmaxf(sv[0][r], sv[1][r]), fmaxf(sv[2][r], sv[3][r]));
#pragma unroll
    for (int m = 1; m < 16; m <<= 1) mx = fmaxf(mx, __shfl_xor(mx, m, 32));
    float sum = 0.f;
#pragma unroll
    for (int tj = 0; tj < 4; ++tj) { sv[tj][r] = __expf(sv[tj][r] - mx); sum += sv[tj][r]; }
#pragma unroll
    for (int m = 1; m < 16; m <<= 1) sum += __shfl_xor(sum, m, 32);
    float inv = 1.0f / sum;
#pragma unroll
    for (int tj = 0; tj < 4; ++tj) sv[tj][r] *= inv;
  }

  // ---- store softmaxed scores (bf16) for A-fragment reload ----
#pragma unroll
  for (int tj = 0; tj < 4; ++tj)
#pragma unroll
    for (int r = 0; r < 8; ++r) {
      int lr = r + (lh << 3);
      sAttn[(wv * 16 + lr) * NTOK + tj * 16 + l15] = f2bf(sv[tj][r]);
    }
  __syncthreads();

  // ---- o = attn @ v ----
  const u16* arow = &sAttn[(wv * 16 + l15) * NTOK];
  v16bf aa0 = ldfrag1(arow + lh * 16);           // K = 0..31
  v16bf aa1 = ldfrag1(arow + 32 + lh * 16); // K = 32..63
#pragma unroll
  for (int cb = 0; cb < 2; ++cb) {
    v16bf bv0 = ldfrag1(&vT[(cb * 16 + l15) * NTOK + lh * 16]);
    v16bf bv1 = ldfrag1(&vT[(cb * 16 + l15) * NTOK + 32 + lh * 16]);
    v8f oa = {};
    oa = WMMA_BF16(aa0, bv0, oa);
    oa = WMMA_BF16(aa1, bv1, oa);
    const float* of = (const float*)&oa;
#pragma unroll
    for (int r = 0; r < 8; ++r) {
      int tok = win * NTOK + wv * 16 + r + (lh << 3);
      int col = h * HD + cb * 16 + l15;
      outH[(size_t)tok * DIMC + col] = f2bf(of[r]);
    }
  }
}

// ---------------------------------------------------------------------------
// Host-side orchestration
// ---------------------------------------------------------------------------
extern "C" void kernel_launch(void* const* d_in, const int* in_sizes, int n_in,
                              void* d_out, int out_size, void* d_ws, size_t ws_size,
                              hipStream_t stream) {
  const float* x     = (const float*)d_in[0];
  const float* g1    = (const float*)d_in[1];
  const float* b1    = (const float*)d_in[2];
  const float* qkvw  = (const float*)d_in[3];
  const float* qkvb  = (const float*)d_in[4];
  const float* projw = (const float*)d_in[5];
  const float* projb = (const float*)d_in[6];
  const float* relb  = (const float*)d_in[7];
  const float* g2    = (const float*)d_in[8];
  const float* b2    = (const float*)d_in[9];
  const float* fc1w  = (const float*)d_in[10];
  const float* fc1b  = (const float*)d_in[11];
  const float* fc2w  = (const float*)d_in[12];
  const float* fc2b  = (const float*)d_in[13];

  char* ws = (char*)d_ws;
  // workspace layout (bytes); BUF_BIG serves qkv / proj-tmp / mlp-hidden
  const size_t OFF_WQKV  = 0;
  const size_t OFF_WPROJ = OFF_WQKV + (size_t)DIMC * QKVN * 2;
  const size_t OFF_WFC1  = OFF_WPROJ + (size_t)DIMC * DIMC * 2;
  const size_t OFF_WFC2  = OFF_WFC1 + (size_t)DIMC * MLPH * 2;
  const size_t OFF_T     = OFF_WFC2 + (size_t)MLPH * DIMC * 2;     // t / t2 bf16
  const size_t OFF_BIG   = OFF_T + (size_t)TOKS * DIMC * 2;
  const size_t OFF_ATTN  = OFF_BIG + (size_t)TOKS * MLPH * 2;      // attn out bf16
  const size_t OFF_X2    = OFF_ATTN + (size_t)TOKS * DIMC * 2;     // x2 fp32

  u16* wqkv  = (u16*)(ws + OFF_WQKV);
  u16* wproj = (u16*)(ws + OFF_WPROJ);
  u16* wfc1  = (u16*)(ws + OFF_WFC1);
  u16* wfc2  = (u16*)(ws + OFF_WFC2);
  u16* tbuf  = (u16*)(ws + OFF_T);
  u16* qkvB  = (u16*)(ws + OFF_BIG);
  float* ptmp = (float*)(ws + OFF_BIG);
  u16* hbuf  = (u16*)(ws + OFF_BIG);
  u16* attnO = (u16*)(ws + OFF_ATTN);
  float* x2  = (float*)(ws + OFF_X2);
  float* out = (float*)d_out;

  // 1) weights -> bf16, transposed to (N x K)
  wcvt_t<<<(DIMC * QKVN + 255) / 256, 256, 0, stream>>>(qkvw, wqkv, DIMC, QKVN);
  wcvt_t<<<(DIMC * DIMC + 255) / 256, 256, 0, stream>>>(projw, wproj, DIMC, DIMC);
  wcvt_t<<<(DIMC * MLPH + 255) / 256, 256, 0, stream>>>(fc1w, wfc1, DIMC, MLPH);
  wcvt_t<<<(MLPH * DIMC + 255) / 256, 256, 0, stream>>>(fc2w, wfc2, MLPH, DIMC);

  // 2) LN1 + shift + window partition
  ln_shift_window<<<TOKS / 8, 256, 0, stream>>>(x, g1, b1, tbuf);

  // 3) QKV GEMM: (131072 x 384) x (384 x 1152) -> bf16
  gemm_bf16<0><<<dim3(QKVN / 64, TOKS / 64), 128, 0, stream>>>(
      tbuf, wqkv, qkvb, nullptr, qkvB, nullptr, TOKS, QKVN, DIMC);

  // 4) windowed attention
  attn_kernel<<<(BB * NWIN) * HEADS, 128, 0, stream>>>(qkvB, relb, attnO);

  // 5) proj GEMM -> fp32 (window-ordered)
  gemm_bf16<1><<<dim3(DIMC / 64, TOKS / 64), 128, 0, stream>>>(
      attnO, wproj, projb, nullptr, nullptr, ptmp, TOKS, DIMC, DIMC);

  // 6) window reverse + unshift + residual + LN2
  residual_ln<<<TOKS / 8, 256, 0, stream>>>(x, ptmp, g2, b2, x2, tbuf);

  // 7) fc1 + GELU -> bf16
  gemm_bf16<2><<<dim3(MLPH / 64, TOKS / 64), 128, 0, stream>>>(
      tbuf, wfc1, fc1b, nullptr, hbuf, nullptr, TOKS, MLPH, DIMC);

  // 8) fc2 + bias + residual -> fp32 output
  gemm_bf16<3><<<dim3(DIMC / 64, TOKS / 64), 128, 0, stream>>>(
      hbuf, wfc2, fc2b, x2, nullptr, out, TOKS, DIMC, MLPH);
}
